// mLSTMCell_27152783245726
// MI455X (gfx1250) — matrix operand order repaired and checked by reading, not compiled, not run on hardware
//
#include <hip/hip_runtime.h>

#define NH 16
#define DH 128
#define S_LEN 2048
#define D_MODEL (NH*DH)
#define EPS 1e-6f
#define QT 16
#define WAVES 8
#define BLOCK (WAVES*32)
#define CHUNK 32

typedef __attribute__((ext_vector_type(16))) _Float16 v16h;
typedef __attribute__((ext_vector_type(8)))  float    v8f;

union Frag16 { v16h v; unsigned int u[8]; _Float16 h[16]; };

// ---------------- Kernel 1: gate preactivations ----------------
__global__ __launch_bounds__(256) void gates_kernel(
    const float* __restrict__ q, const float* __restrict__ k, const float* __restrict__ v,
    const float* __restrict__ ikern, const float* __restrict__ ibias,
    const float* __restrict__ fkern, const float* __restrict__ fbias,
    float* __restrict__ ig, float* __restrict__ lf)
{
  const int s = blockIdx.x;
  const int tid = threadIdx.x;
  float accI[NH], accF[NH];
  #pragma unroll
  for (int h = 0; h < NH; ++h) { accI[h] = 0.f; accF[h] = 0.f; }
  for (int c = 0; c < 24; ++c) {
    int dg = tid + 256 * c;
    float x;
    if (dg < D_MODEL)            x = q[s * D_MODEL + dg];
    else if (dg < 2 * D_MODEL)   x = k[s * D_MODEL + dg - D_MODEL];
    else                         x = v[s * D_MODEL + dg - 2 * D_MODEL];
    const float* ik = ikern + (size_t)dg * NH;
    const float* fk = fkern + (size_t)dg * NH;
    #pragma unroll
    for (int h = 0; h < NH; ++h) {
      accI[h] = fmaf(x, ik[h], accI[h]);
      accF[h] = fmaf(x, fk[h], accF[h]);
    }
  }
  #pragma unroll
  for (int h = 0; h < NH; ++h) {
    #pragma unroll
    for (int o = 16; o >= 1; o >>= 1) {
      accI[h] += __shfl_xor(accI[h], o);
      accF[h] += __shfl_xor(accF[h], o);
    }
  }
  __shared__ float redI[8][NH];
  __shared__ float redF[8][NH];
  int wave = tid >> 5, lane = tid & 31;
  if (lane == 0) {
    #pragma unroll
    for (int h = 0; h < NH; ++h) { redI[wave][h] = accI[h]; redF[wave][h] = accF[h]; }
  }
  __syncthreads();
  if (tid < NH) {
    float si = ibias[tid], sf = fbias[tid];
    #pragma unroll
    for (int w = 0; w < 8; ++w) { si += redI[w][tid]; sf += redF[w][tid]; }
    ig[tid * S_LEN + s] = si;
    // log_sigmoid(sf) = min(sf,0) - log1p(exp(-|sf|))
    lf[tid * S_LEN + s] = fminf(sf, 0.f) - log1pf(__expf(-fabsf(sf)));
  }
}

// ---------------- Kernel 2: per-head cumulative sum of log-forget ----------------
__global__ __launch_bounds__(512) void cumsum_kernel(
    const float* __restrict__ lf, float* __restrict__ cum)
{
  int h = threadIdx.x >> 5;
  int lane = threadIdx.x & 31;
  float carry = 0.f;
  if (lane == 0) cum[h * (S_LEN + 1)] = 0.f;
  for (int base = 0; base < S_LEN; base += 32) {
    float x = lf[h * S_LEN + base + lane];
    #pragma unroll
    for (int o = 1; o < 32; o <<= 1) {
      float y = __shfl_up(x, o);
      if (lane >= (int)o) x += y;
    }
    cum[h * (S_LEN + 1) + base + lane + 1] = carry + x;
    carry += __shfl(x, 31);
  }
}

// ---------------- Kernel 3: flash-style stabilized mLSTM attention ----------------
__global__ __launch_bounds__(BLOCK) void mlstm_attn_kernel(
    const float* __restrict__ q, const float* __restrict__ k, const float* __restrict__ v,
    const float* __restrict__ ig, const float* __restrict__ cum,
    const float* __restrict__ onw, float* __restrict__ out)
{
  __shared__ _Float16 k_lds[CHUNK * DH];        // (t, d) row-major
  __shared__ _Float16 v_lds[DH * CHUNK];        // (d, t) transposed
  __shared__ _Float16 p_lds[WAVES * QT * CHUNK];// per-wave P transpose bounce

  const int h = blockIdx.y;
  const int qbase = blockIdx.x * (WAVES * QT);
  const int tid = threadIdx.x;
  const int wave = tid >> 5;
  const int lane = tid & 31;
  const int half = lane >> 4;    // 0/1 lane group
  const int l16  = lane & 15;
  const int qw = qbase + wave * QT;

  const float inv_sqrt_dh = 0.08838834764831845f; // 1/sqrt(128)

  // Q tile as 4 A-fragments (K-dim slices of 32), documented 16-bit A layout
  Frag16 qa[4];
  #pragma unroll
  for (int f = 0; f < 4; ++f) {
    #pragma unroll
    for (int p = 0; p < 8; ++p) {
      int kk = f * 32 + (p >> 2) * 16 + half * 8 + (p & 3) * 2;
      const float* src = q + (size_t)(qw + l16) * D_MODEL + h * DH + kk;
      float2 x = *(const float2*)src;
      qa[f].h[2 * p]     = (_Float16)x.x;
      qa[f].h[2 * p + 1] = (_Float16)x.y;
    }
  }

  float cq[8];
  #pragma unroll
  for (int r = 0; r < 8; ++r)
    cq[r] = cum[h * (S_LEN + 1) + qw + r + half * 8 + 1];

  const v8f vzero = {0.f, 0.f, 0.f, 0.f, 0.f, 0.f, 0.f, 0.f};
  v8f oacc[8];
  #pragma unroll
  for (int n = 0; n < 8; ++n) oacc[n] = vzero;
  float m_r[8], l_r[8];
  #pragma unroll
  for (int r = 0; r < 8; ++r) { m_r[r] = -__builtin_inff(); l_r[r] = 0.f; }

  const int nchunks = (qbase + WAVES * QT) / CHUNK;
  for (int ci = 0; ci < nchunks; ++ci) {
    const int t0 = ci * CHUNK;
    // ---- cooperative K/V chunk load (256 threads), f32 -> f16, V transposed ----
    {
      int row = tid >> 3;             // 0..31
      int cg  = (tid & 7) * 16;       // column group base
      const float* kp = k + (size_t)(t0 + row) * D_MODEL + h * DH + cg;
      const float* vp = v + (size_t)(t0 + row) * D_MODEL + h * DH + cg;
      #pragma unroll
      for (int j = 0; j < 4; ++j) {
        int col = cg + j * 4;
        float4 kx = *(const float4*)(kp + j * 4);
        _Float16* kd = &k_lds[row * DH + col];
        kd[0] = (_Float16)kx.x; kd[1] = (_Float16)kx.y;
        kd[2] = (_Float16)kx.z; kd[3] = (_Float16)kx.w;
        float4 vx = *(const float4*)(vp + j * 4);
        v_lds[(col + 0) * CHUNK + row] = (_Float16)vx.x;
        v_lds[(col + 1) * CHUNK + row] = (_Float16)vx.y;
        v_lds[(col + 2) * CHUNK + row] = (_Float16)vx.z;
        v_lds[(col + 3) * CHUNK + row] = (_Float16)vx.w;
      }
      // prefetch next chunk rows into cache while this chunk computes
      if (t0 + CHUNK + row < S_LEN) {
        __builtin_prefetch(kp + (size_t)CHUNK * D_MODEL, 0, 3);
        __builtin_prefetch(vp + (size_t)CHUNK * D_MODEL, 0, 3);
      }
    }
    __syncthreads();

    // wave-uniform causal skip: this wave's queries all precede the chunk
    if (t0 <= qw + QT - 1) {
      // ---- QK^T : two 16x16 score tiles, reduction over d=128 ----
      v8f sA = vzero, sB = vzero;
      #pragma unroll
      for (int f = 0; f < 4; ++f) {
        Frag16 bA, bB;
        #pragma unroll
        for (int p = 0; p < 8; ++p) {
          int d = f * 32 + (p >> 2) * 16 + half * 8 + (p & 3) * 2;
          bA.u[p] = *(const unsigned int*)&k_lds[l16 * DH + d];
          bB.u[p] = *(const unsigned int*)&k_lds[(16 + l16) * DH + d];
        }
        sA = __builtin_amdgcn_wmma_f32_16x16x32_f16(false, qa[f].v, false, bA.v, (short)0, sA, false, false);
        sB = __builtin_amdgcn_wmma_f32_16x16x32_f16(false, qa[f].v, false, bB.v, (short)0, sB, false, false);
      }

      // ---- decay gating + online stabilization ----
      float ctA = cum[h * (S_LEN + 1) + t0 + l16 + 1];
      float itA = ig[h * S_LEN + t0 + l16];
      float ctB = cum[h * (S_LEN + 1) + t0 + 16 + l16 + 1];
      float itB = ig[h * S_LEN + t0 + 16 + l16];

      float pA[8], pB[8];
      #pragma unroll
      for (int r = 0; r < 8; ++r) {
        int qg = qw + r + half * 8;
        bool vA = (t0 + l16)      <= qg;
        bool vB = (t0 + 16 + l16) <= qg;
        float ldA = vA ? (cq[r] - ctA + itA) : -__builtin_inff();
        float ldB = vB ? (cq[r] - ctB + itB) : -__builtin_inff();
        float cm = fmaxf(ldA, ldB);
        #pragma unroll
        for (int o = 1; o < 16; o <<= 1) cm = fmaxf(cm, __shfl_xor(cm, o));
        float newm = fmaxf(m_r[r], cm);
        float scale = __expf(m_r[r] - newm);
        float eA = vA ? sA[r] * inv_sqrt_dh * __expf(ldA - newm) : 0.f;
        float eB = vB ? sB[r] * inv_sqrt_dh * __expf(ldB - newm) : 0.f;
        float rsum = eA + eB;
        #pragma unroll
        for (int o = 1; o < 16; o <<= 1) rsum += __shfl_xor(rsum, o);
        l_r[r] = l_r[r] * scale + rsum;
        m_r[r] = newm;
        #pragma unroll
        for (int n = 0; n < 8; ++n) oacc[n][r] *= scale;
        pA[r] = eA; pB[r] = eB;
      }

      // ---- C-layout -> A-layout transpose of P through LDS, then P @ V ----
      {
        _Float16* pw = &p_lds[wave * QT * CHUNK];
        #pragma unroll
        for (int r = 0; r < 8; ++r) {
          int mrow = r + half * 8;
          pw[mrow * CHUNK + l16]      = (_Float16)pA[r];
          pw[mrow * CHUNK + 16 + l16] = (_Float16)pB[r];
        }
        Frag16 pfr;
        #pragma unroll
        for (int p = 0; p < 8; ++p) {
          int kk = (p >> 2) * 16 + half * 8 + (p & 3) * 2;
          pfr.u[p] = *(const unsigned int*)&pw[l16 * CHUNK + kk];
        }
        #pragma unroll
        for (int n = 0; n < 8; ++n) {
          Frag16 bv;
          #pragma unroll
          for (int p = 0; p < 8; ++p) {
            int t = (p >> 2) * 16 + half * 8 + (p & 3) * 2;
            int d = n * 16 + l16;
            bv.u[p] = *(const unsigned int*)&v_lds[d * CHUNK + t];
          }
          oacc[n] = __builtin_amdgcn_wmma_f32_16x16x32_f16(false, pfr.v, false, bv.v, (short)0, oacc[n], false, false);
        }
      }
    }
    __syncthreads();
  }

  // ---- finalize: stabilized normalizer + per-head layernorm ----
  float inv_nrm[8];
  #pragma unroll
  for (int r = 0; r < 8; ++r)
    inv_nrm[r] = 1.f / (fmaxf(fabsf(l_r[r]), __expf(-m_r[r])) + EPS);

  float mean[8];
  #pragma unroll
  for (int r = 0; r < 8; ++r) {
    float s = 0.f;
    #pragma unroll
    for (int n = 0; n < 8; ++n) { oacc[n][r] *= inv_nrm[r]; s += oacc[n][r]; }
    #pragma unroll
    for (int o = 1; o < 16; o <<= 1) s += __shfl_xor(s, o);
    mean[r] = s * (1.f / 128.f);
  }
  #pragma unroll
  for (int r = 0; r < 8; ++r) {
    float vv = 0.f;
    #pragma unroll
    for (int n = 0; n < 8; ++n) { float dlt = oacc[n][r] - mean[r]; vv += dlt * dlt; }
    #pragma unroll
    for (int o = 1; o < 16; o <<= 1) vv += __shfl_xor(vv, o);
    float rstd = rsqrtf(vv * (1.f / 128.f) + EPS);
    int qg = qw + r + half * 8;
    #pragma unroll
    for (int n = 0; n < 8; ++n) {
      int d = n * 16 + l16;
      float w = onw[h * DH + d];
      out[(size_t)qg * D_MODEL + h * DH + d] = (oacc[n][r] - mean[r]) * rstd * w;
    }
  }
}

extern "C" void kernel_launch(void* const* d_in, const int* in_sizes, int n_in,
                              void* d_out, int out_size, void* d_ws, size_t ws_size,
                              hipStream_t stream)
{
  const float* q   = (const float*)d_in[0];
  const float* k   = (const float*)d_in[1];
  const float* v   = (const float*)d_in[2];
  const float* ik  = (const float*)d_in[3];
  const float* ib  = (const float*)d_in[4];
  const float* fk  = (const float*)d_in[5];
  const float* fb  = (const float*)d_in[6];
  const float* onw = (const float*)d_in[7];
  float* out = (float*)d_out;

  float* ig  = (float*)d_ws;          // NH*S
  float* lf  = ig + NH * S_LEN;       // NH*S
  float* cum = lf + NH * S_LEN;       // NH*(S+1)

  gates_kernel<<<S_LEN, 256, 0, stream>>>(q, k, v, ik, ib, fk, fb, ig, lf);
  cumsum_kernel<<<1, 512, 0, stream>>>(lf, cum);
  dim3 grid(S_LEN / (WAVES * QT), NH);
  mlstm_attn_kernel<<<grid, BLOCK, 0, stream>>>(q, k, v, ig, cum, onw, out);
}
